// Attention_27650999452015
// MI455X (gfx1250) — matrix operand order repaired and checked by reading, not compiled, not run on hardware
//
#include <hip/hip_runtime.h>
#include <math.h>

// ---------------- problem constants ----------------
#define BB      512
#define TT      64
#define INP     512
#define HIDN    512
#define NCLS    96
#define NSTEPS  27
#define KX      (INP + NCLS)   // 608 = 19*32
#define G4      (4 * HIDN)     // 2048

// ---------------- WMMA types ----------------
typedef __attribute__((ext_vector_type(16))) __bf16 v16bf;
typedef __attribute__((ext_vector_type(8)))  float  v8f;

union FragU { v16bf v; uint4 u[2]; };

static __device__ __forceinline__ __bf16 f2bf(float f) {
    unsigned int u = __builtin_bit_cast(unsigned int, f);
    unsigned int r = (u + 0x7FFFu + ((u >> 16) & 1u)) >> 16;
    unsigned short s = (unsigned short)r;
    return __builtin_bit_cast(__bf16, s);
}
static __device__ __forceinline__ float bf2f_bits(unsigned short s) {
    unsigned int u = ((unsigned int)s) << 16;
    return __builtin_bit_cast(float, u);
}
static __device__ __forceinline__ float sigmoidf_(float x) {
    return 1.0f / (1.0f + expf(-x));
}

// ---------------- CDNA5 async global->LDS staging (guarded) ----------------
#if defined(__has_builtin)
#if __has_builtin(__builtin_amdgcn_global_load_async_to_lds_b128)
#define USE_ASYNC_LDS 1
#endif
#endif

// Builtin prototype (from hipcc diagnostic): param0 is
// 'int __attribute__((vector_size(16))) __device__ *'  (global AS1 int4*).
typedef int v4i_gcc __attribute__((vector_size(16)));
typedef __attribute__((address_space(1))) v4i_gcc gv4i_t;
typedef __attribute__((address_space(3))) v4i_gcc lv4i_t;

static __device__ __forceinline__ void stage_cp16(const __bf16* g, __bf16* l) {
#if USE_ASYNC_LDS
    // AS3 pointers are 32-bit; generic LDS pointers keep the LDS offset in addr[31:0].
    __builtin_amdgcn_global_load_async_to_lds_b128(
        (gv4i_t*)(__UINTPTR_TYPE__)g,
        (lv4i_t*)(unsigned int)(__UINTPTR_TYPE__)l,
        0, 0);
#else
    *(uint4*)l = *(const uint4*)g;
#endif
}

static __device__ __forceinline__ void stage_fence() {
#if USE_ASYNC_LDS
#if __has_builtin(__builtin_amdgcn_s_wait_asynccnt)
    __builtin_amdgcn_s_wait_asynccnt(0);
#else
    asm volatile("s_wait_asynccnt 0x0" ::: "memory");
#endif
#endif
}

// A fragment: 16x32 bf16, row-major source A[row, k], lda elements.
// Contiguous k order used here (same as load_b_frag): lanes 0-15 = M rows; elems 0..15 -> K=kb..kb+15,
// kb = k0 + (lane>=16 ? 16 : 0).
static __device__ __forceinline__ v16bf load_a_frag(const __bf16* __restrict__ A, int lda,
                                                    int m0, int k0, int lane) {
    int row   = m0 + (lane & 15);
    int kbase = k0 + ((lane >> 4) << 4);
    const __bf16* p = A + (size_t)row * lda + kbase;
    FragU f;
    f.u[0] = *(const uint4*)(p);
    f.u[1] = *(const uint4*)(p + 8);
    return f.v;
}

// B fragment: 32x16 bf16 from W[n, k] (row-major, ldw elements) -> B = W^T slice.
// Contiguous k order used here (same as load_a_frag): lanes 0-15 hold K=k0..k0+15, lanes 16-31 hold K=k0+16..k0+31; N = lane&15.
static __device__ __forceinline__ v16bf load_b_frag(const __bf16* __restrict__ W, int ldw,
                                                    int n0, int k0, int lane) {
    int n     = n0 + (lane & 15);
    int kbase = k0 + ((lane >> 4) << 4);
    const __bf16* p = W + (size_t)n * ldw + kbase;
    FragU f;
    f.u[0] = *(const uint4*)(p);
    f.u[1] = *(const uint4*)(p + 8);
    return f.v;
}

// ---------------- generic GEMM (direct global): out = A * W^T (+bias) -----
// One wave -> 16x64 output tile. Used for the small latency-critical GEMMs.
template <bool OUT_BF16>
__global__ void wmma_gemm_bf16(const __bf16* __restrict__ A, int lda,
                               const __bf16* __restrict__ W, int ldw,
                               const float* __restrict__ bias,
                               void* __restrict__ out, int ldo,
                               int M, int N, int K) {
    int wave = (blockIdx.x * blockDim.x + threadIdx.x) >> 5;
    int lane = threadIdx.x & 31;
    int n4   = N >> 6;
    int mt   = wave / n4;
    int nt   = (wave - mt * n4) << 2;
    if (mt * 16 >= M) return;

    v8f acc[4] = {};
    for (int k0 = 0; k0 < K; k0 += 32) {
        v16bf a = load_a_frag(A, lda, mt * 16, k0, lane);
#pragma unroll
        for (int j = 0; j < 4; ++j) {
            v16bf b = load_b_frag(W, ldw, (nt + j) * 16, k0, lane);
            acc[j] = __builtin_amdgcn_wmma_f32_16x16x32_bf16(
                false, a, false, b, (short)0, acc[j], false, false);
        }
    }

    int laneHi = lane >> 4;
#pragma unroll
    for (int j = 0; j < 4; ++j) {
        int n = (nt + j) * 16 + (lane & 15);
        float bv = bias ? bias[n] : 0.0f;
#pragma unroll
        for (int r = 0; r < 8; ++r) {
            int m = mt * 16 + r + (laneHi << 3);
            float v = acc[j][r] + bv;
            if (OUT_BF16) ((__bf16*)out)[(size_t)m * ldo + n] = f2bf(v);
            else          ((float*)out)[(size_t)m * ldo + n]  = v;
        }
    }
}

// ---------------- staged GEMM: B tile in LDS via async copy, 4-way shared --
// Block = 4 waves = 64(M) x 64(N) tile. B tile (64x32 bf16) staged into LDS
// once per K-step (double buffered, rows padded to 40 elems = 20 banks ->
// conflict-free ds_load_b128), shared by all 4 waves. A loaded direct with
// register double-buffering.
#define BROW 40
template <bool OUT_BF16>
__global__ void wmma_gemm_bf16_staged(const __bf16* __restrict__ A, int lda,
                                      const __bf16* __restrict__ W, int ldw,
                                      const float* __restrict__ bias,
                                      void* __restrict__ out, int ldo,
                                      int M, int N, int K) {
    __shared__ __align__(16) __bf16 sB[2][64 * BROW];

    int tid  = threadIdx.x;            // 128 threads = 4 waves
    int lane = tid & 31;
    int wv   = tid >> 5;
    int nbN  = N >> 6;
    int bm   = blockIdx.x / nbN;
    int bn   = blockIdx.x - bm * nbN;
    int m0   = bm * 64 + wv * 16;
    int n0   = bn * 64;

    // stage one 64x32 B tile: 256 16B chunks, 2 per thread
    auto stage = [&](int buf, int k0) {
#pragma unroll
        for (int rep = 0; rep < 2; ++rep) {
            int q = tid + rep * 128;
            int r = q >> 2, c = q & 3;
            stage_cp16(W + (size_t)(n0 + r) * ldw + k0 + c * 8,
                       &sB[buf][r * BROW + c * 8]);
        }
    };

    stage(0, 0);
    v16bf a_cur = load_a_frag(A, lda, m0, 0, lane);
    stage_fence();
    __syncthreads();

    v8f acc[4] = {};
    int ksteps = K >> 5;
    for (int ks = 0; ks < ksteps; ++ks) {
        int  cur  = ks & 1;
        bool more = (ks + 1) < ksteps;
        if (more) stage(cur ^ 1, (ks + 1) << 5);
        v16bf a_next = a_cur;
        if (more) a_next = load_a_frag(A, lda, m0, (ks + 1) << 5, lane);

        int kb = (lane >> 4) << 4;
#pragma unroll
        for (int j = 0; j < 4; ++j) {
            const __bf16* p = &sB[cur][(j * 16 + (lane & 15)) * BROW + kb];
            FragU f;
            f.u[0] = *(const uint4*)(p);
            f.u[1] = *(const uint4*)(p + 8);
            acc[j] = __builtin_amdgcn_wmma_f32_16x16x32_bf16(
                false, a_cur, false, f.v, (short)0, acc[j], false, false);
        }
        a_cur = a_next;
        if (more) { stage_fence(); __syncthreads(); }
    }

    int laneHi = lane >> 4;
#pragma unroll
    for (int j = 0; j < 4; ++j) {
        int n = n0 + j * 16 + (lane & 15);
        float bv = bias ? bias[n] : 0.0f;
#pragma unroll
        for (int r = 0; r < 8; ++r) {
            int m = m0 + r + (laneHi << 3);
            float v = acc[j][r] + bv;
            if (OUT_BF16) ((__bf16*)out)[(size_t)m * ldo + n] = f2bf(v);
            else          ((float*)out)[(size_t)m * ldo + n]  = v;
        }
    }
}

// ---------------- gates GEMM: gates[B,2048] = x*W_ih^T + h*W_hh^T + biases -
__global__ void wmma_gates_gemm(const __bf16* __restrict__ X,   // [B, 608]
                                const __bf16* __restrict__ Wih, // [2048, 608]
                                const __bf16* __restrict__ H,   // [B, 512]
                                const __bf16* __restrict__ Whh, // [2048, 512]
                                const float* __restrict__ bih,
                                const float* __restrict__ bhh,
                                float* __restrict__ out) {
    const int n4 = G4 >> 6; // 32
    int wave = (blockIdx.x * blockDim.x + threadIdx.x) >> 5;
    int lane = threadIdx.x & 31;
    int mt = wave / n4;
    int nt = (wave - mt * n4) << 2;
    if (mt * 16 >= BB) return;

    v8f acc[4] = {};
    for (int k0 = 0; k0 < KX; k0 += 32) {
        v16bf a = load_a_frag(X, KX, mt * 16, k0, lane);
#pragma unroll
        for (int j = 0; j < 4; ++j) {
            v16bf b = load_b_frag(Wih, KX, (nt + j) * 16, k0, lane);
            acc[j] = __builtin_amdgcn_wmma_f32_16x16x32_bf16(
                false, a, false, b, (short)0, acc[j], false, false);
        }
    }
    for (int k0 = 0; k0 < HIDN; k0 += 32) {
        v16bf a = load_a_frag(H, HIDN, mt * 16, k0, lane);
#pragma unroll
        for (int j = 0; j < 4; ++j) {
            v16bf b = load_b_frag(Whh, HIDN, (nt + j) * 16, k0, lane);
            acc[j] = __builtin_amdgcn_wmma_f32_16x16x32_bf16(
                false, a, false, b, (short)0, acc[j], false, false);
        }
    }

    int laneHi = lane >> 4;
#pragma unroll
    for (int j = 0; j < 4; ++j) {
        int n = (nt + j) * 16 + (lane & 15);
        float bv = bih[n] + bhh[n];
#pragma unroll
        for (int r = 0; r < 8; ++r) {
            int m = mt * 16 + r + (laneHi << 3);
            out[(size_t)m * G4 + n] = acc[j][r] + bv;
        }
    }
}

// ---------------- final projection with permuted store --------------------
// hiddens stored [NSTEPS, B, HID]; out is [B, NSTEPS, NCLS].
__global__ void wmma_final_gemm(const __bf16* __restrict__ Hd,  // [NSTEPS*B, 512]
                                const __bf16* __restrict__ Wg,  // [96, 512]
                                const float* __restrict__ bg,   // [96]
                                float* __restrict__ out) {
    const int ntiles = NCLS / 16; // 6
    int wave = (blockIdx.x * blockDim.x + threadIdx.x) >> 5;
    int lane = threadIdx.x & 31;
    int mt = wave / ntiles;
    int nt = wave - mt * ntiles;
    if (mt * 16 >= NSTEPS * BB) return;

    v8f acc = {};
    for (int k0 = 0; k0 < HIDN; k0 += 32) {
        v16bf a = load_a_frag(Hd, HIDN, mt * 16, k0, lane);
        v16bf b = load_b_frag(Wg, HIDN, nt * 16, k0, lane);
        acc = __builtin_amdgcn_wmma_f32_16x16x32_bf16(
            false, a, false, b, (short)0, acc, false, false);
    }

    int laneHi = lane >> 4;
    int n = nt * 16 + (lane & 15);
    float bv = bg[n];
#pragma unroll
    for (int r = 0; r < 8; ++r) {
        int m = mt * 16 + r + (laneHi << 3);
        int s = m >> 9;        // m / B
        int b = m & (BB - 1);  // m % B
        out[(size_t)b * (NSTEPS * NCLS) + (size_t)s * NCLS + n] = acc[r] + bv;
    }
}

// ---------------- fused attention: score -> softmax -> context -> x -------
__global__ void attention_kernel(const float* __restrict__ prev,   // [B, 512]
                                 const __bf16* __restrict__ bhp,   // [B, T, 512]
                                 const float* __restrict__ bH,     // [B, T, 512] fp32
                                 const float* __restrict__ wscore, // [512]
                                 const int* __restrict__ text,     // [B, NSTEPS]
                                 int step,
                                 __bf16* __restrict__ x) {         // [B, 608]
    __shared__ float sprev[HIDN];
    __shared__ float sws[HIDN];
    __shared__ float se[TT];
    __shared__ float salpha[TT];
    __shared__ float sred[2];

    int b   = blockIdx.x;
    int tid = threadIdx.x;            // 256 threads = 8 waves
    for (int i = tid; i < HIDN; i += 256) { sprev[i] = prev[(size_t)b * HIDN + i]; sws[i] = wscore[i]; }
    __syncthreads();

    int wv = tid >> 5, ln = tid & 31;
    const unsigned short* bp = (const unsigned short*)(bhp + (size_t)b * TT * HIDN);
#pragma unroll
    for (int j = 0; j < 8; ++j) {
        int t = wv * 8 + j;
        float p = 0.0f;
        for (int h = ln; h < HIDN; h += 32)
            p += tanhf(bf2f_bits(bp[t * HIDN + h]) + sprev[h]) * sws[h];
#pragma unroll
        for (int off = 16; off > 0; off >>= 1) p += __shfl_xor(p, off, 32);
        if (ln == 0) se[t] = p;
    }
    __syncthreads();

    if (tid < 32) {
        float m = fmaxf(se[tid], se[tid + 32]);
#pragma unroll
        for (int off = 16; off > 0; off >>= 1) m = fmaxf(m, __shfl_xor(m, off, 32));
        if (tid == 0) sred[0] = m;
    }
    __syncthreads();
    if (tid < TT) salpha[tid] = expf(se[tid] - sred[0]);
    __syncthreads();
    if (tid < 32) {
        float s = salpha[tid] + salpha[tid + 32];
#pragma unroll
        for (int off = 16; off > 0; off >>= 1) s += __shfl_xor(s, off, 32);
        if (tid == 0) sred[1] = s;
    }
    __syncthreads();
    float inv = 1.0f / sred[1];

    const float* ph = bH + (size_t)b * TT * INP;
    for (int i = tid; i < INP; i += 256) {
        float acc = 0.0f;
#pragma unroll 8
        for (int t = 0; t < TT; ++t) acc += salpha[t] * ph[(size_t)t * INP + i];
        x[(size_t)b * KX + i] = f2bf(acc * inv);
    }
    int ch = text[b * NSTEPS + step];
    for (int c = tid; c < NCLS; c += 256)
        x[(size_t)b * KX + INP + c] = f2bf(c == ch ? 1.0f : 0.0f);
}

// ---------------- LSTM elementwise (PyTorch gate order i,f,g,o) -----------
__global__ void lstm_cell_kernel(const float* __restrict__ gates, // [B, 2048]
                                 float* __restrict__ c,           // [B, 512]
                                 __bf16* __restrict__ h,          // [B, 512]
                                 __bf16* __restrict__ hid_s) {    // [B, 512]
    int idx = blockIdx.x * blockDim.x + threadIdx.x;
    if (idx >= BB * HIDN) return;
    int b = idx >> 9, j = idx & (HIDN - 1);
    const float* g = gates + (size_t)b * G4;
    float ig = sigmoidf_(g[j]);
    float fg = sigmoidf_(g[HIDN + j]);
    float gg = tanhf(g[2 * HIDN + j]);
    float og = sigmoidf_(g[3 * HIDN + j]);
    float cn = fg * c[idx] + ig * gg;
    float hn = og * tanhf(cn);
    c[idx] = cn;
    __bf16 hb = f2bf(hn);
    h[idx] = hb;
    hid_s[idx] = hb;
}

// ---------------- utility kernels -----------------------------------------
__global__ void cvt_f32_to_bf16(__bf16* __restrict__ dst, const float* __restrict__ src, int n) {
    for (int i = blockIdx.x * blockDim.x + threadIdx.x; i < n; i += gridDim.x * blockDim.x)
        dst[i] = f2bf(src[i]);
}
__global__ void zero_u16(unsigned short* p, int n) {
    for (int i = blockIdx.x * blockDim.x + threadIdx.x; i < n; i += gridDim.x * blockDim.x) p[i] = 0;
}
__global__ void zero_f32k(float* p, int n) {
    for (int i = blockIdx.x * blockDim.x + threadIdx.x; i < n; i += gridDim.x * blockDim.x) p[i] = 0.0f;
}

// ---------------- host side ------------------------------------------------
static inline size_t align256(size_t x) { return (x + 255) & ~(size_t)255; }

extern "C" void kernel_launch(void* const* d_in, const int* in_sizes, int n_in,
                              void* d_out, int out_size, void* d_ws, size_t ws_size,
                              hipStream_t stream) {
    const float* batch_H = (const float*)d_in[0];
    const int*   text    = (const int*)d_in[1];
    const float* W_i2h   = (const float*)d_in[2];
    const float* W_h2h   = (const float*)d_in[3];
    const float* b_h2h   = (const float*)d_in[4];
    const float* W_score = (const float*)d_in[5];
    const float* W_ih    = (const float*)d_in[6];
    const float* b_ih    = (const float*)d_in[7];
    const float* W_hh    = (const float*)d_in[8];
    const float* b_hh    = (const float*)d_in[9];
    const float* W_gen   = (const float*)d_in[10];
    const float* b_gen   = (const float*)d_in[11];
    float* out = (float*)d_out;

    // workspace carve-up
    char* w = (char*)d_ws;
    size_t off = 0;
    auto alloc = [&](size_t bytes) { void* p = w + off; off = align256(off + bytes); return p; };
    __bf16* bh_bf   = (__bf16*)alloc((size_t)BB * TT * INP * 2);
    __bf16* bhp_bf  = (__bf16*)alloc((size_t)BB * TT * HIDN * 2);
    __bf16* Wi2h_bf = (__bf16*)alloc((size_t)HIDN * INP * 2);
    __bf16* Wh2h_bf = (__bf16*)alloc((size_t)HIDN * HIDN * 2);
    __bf16* Wih_bf  = (__bf16*)alloc((size_t)G4 * KX * 2);
    __bf16* Whh_bf  = (__bf16*)alloc((size_t)G4 * HIDN * 2);
    __bf16* Wgen_bf = (__bf16*)alloc((size_t)NCLS * HIDN * 2);
    __bf16* h_bf    = (__bf16*)alloc((size_t)BB * HIDN * 2);
    float*  c_f     = (float*)alloc((size_t)BB * HIDN * 4);
    float*  prev_f  = (float*)alloc((size_t)BB * HIDN * 4);
    __bf16* x_bf    = (__bf16*)alloc((size_t)BB * KX * 2);
    float*  gates_f = (float*)alloc((size_t)BB * G4 * 4);
    __bf16* hid_bf  = (__bf16*)alloc((size_t)NSTEPS * BB * HIDN * 2);
    (void)ws_size; (void)n_in; (void)in_sizes; (void)out_size;

    // 1) precision conversion
    cvt_f32_to_bf16<<<4096, 256, 0, stream>>>(bh_bf, batch_H, BB * TT * INP);
    cvt_f32_to_bf16<<<512, 256, 0, stream>>>(Wi2h_bf, W_i2h, HIDN * INP);
    cvt_f32_to_bf16<<<512, 256, 0, stream>>>(Wh2h_bf, W_h2h, HIDN * HIDN);
    cvt_f32_to_bf16<<<2048, 256, 0, stream>>>(Wih_bf, W_ih, G4 * KX);
    cvt_f32_to_bf16<<<2048, 256, 0, stream>>>(Whh_bf, W_hh, G4 * HIDN);
    cvt_f32_to_bf16<<<96, 256, 0, stream>>>(Wgen_bf, W_gen, NCLS * HIDN);

    // 2) zero initial state
    zero_u16<<<512, 256, 0, stream>>>((unsigned short*)h_bf, BB * HIDN);
    zero_f32k<<<512, 256, 0, stream>>>(c_f, BB * HIDN);

    // 3) batch_H_proj = batch_H @ W_i2h^T  (M=32768, N=512, K=512) -> bf16
    //    staged variant: B tile shared via (async) LDS staging.
    {
        int blocks = (BB * TT / 64) * (HIDN / 64);   // 512 * 8 = 4096
        wmma_gemm_bf16_staged<true><<<blocks, 128, 0, stream>>>(
            bh_bf, INP, Wi2h_bf, INP, nullptr, (void*)bhp_bf, HIDN,
            BB * TT, HIDN, INP);
    }

    // 4) 27 recurrent steps
    for (int s = 0; s < NSTEPS; ++s) {
        {
            int waves = (BB / 16) * (HIDN / 64);    // 256
            wmma_gemm_bf16<false><<<waves / 4, 128, 0, stream>>>(
                h_bf, HIDN, Wh2h_bf, HIDN, b_h2h, (void*)prev_f, HIDN,
                BB, HIDN, HIDN);
        }
        attention_kernel<<<BB, 256, 0, stream>>>(prev_f, bhp_bf, batch_H, W_score,
                                                 text, s, x_bf);
        {
            int waves = (BB / 16) * (G4 / 64);      // 1024
            wmma_gates_gemm<<<waves / 4, 128, 0, stream>>>(
                x_bf, Wih_bf, h_bf, Whh_bf, b_ih, b_hh, gates_f);
        }
        lstm_cell_kernel<<<(BB * HIDN) / 256, 256, 0, stream>>>(
            gates_f, c_f, h_bf, hid_bf + (size_t)s * BB * HIDN);
    }

    // 5) probs = hiddens @ W_gen^T + b_gen, permuted to [B, NSTEPS, NCLS]
    {
        int waves = (NSTEPS * BB / 16) * (NCLS / 16);   // 5184
        wmma_final_gemm<<<waves / 4, 128, 0, stream>>>(hid_bf, Wgen_bf, b_gen, out);
    }
}
